// FusedQKVAttention_Simulated_50508815401127
// MI455X (gfx1250) — matrix operand order repaired and checked by reading, not compiled, not run on hardware
//
#include <hip/hip_runtime.h>

#define DIM  2048
#define NSEQ 2048
#define NB   4
#define NH   16
#define DH   128

typedef __attribute__((ext_vector_type(16))) __bf16 v16bf;
typedef __attribute__((ext_vector_type(8)))  float  v8f;
typedef int v4i __attribute__((vector_size(16)));

union FragAB { v16bf v; unsigned u[8]; unsigned short h[16]; };
union FragC  { v8f   v; float f[8]; };

__device__ __forceinline__ unsigned short f2bf(float f) {
  unsigned u = __builtin_bit_cast(unsigned, f);
  u += 0x7FFFu + ((u >> 16) & 1u);          // round-to-nearest-even
  return (unsigned short)(u >> 16);
}

// A-matrix (16x32 bf16) K-index for VGPR v, lane-half h (contiguous: k = 16h + 2v)
__device__ __forceinline__ int kmapA(int v, int h) {
  return 2 * v + 16 * h;
}

__device__ __forceinline__ v8f bf16_wmma(v16bf a, v16bf b, v8f c) {
  return __builtin_amdgcn_wmma_f32_16x16x32_bf16(false, a, false, b,
                                                 (short)0, c, false, false);
}

#if defined(__has_builtin)
#if __has_builtin(__builtin_amdgcn_global_load_async_to_lds_b128) && \
    __has_builtin(__builtin_amdgcn_s_wait_asynccnt)
#define HAVE_ASYNC_LDS 1
#endif
#endif

__device__ __forceinline__ void cp16(const void* g, void* l) {
#ifdef HAVE_ASYNC_LDS
  __builtin_amdgcn_global_load_async_to_lds_b128(
      (__attribute__((address_space(1))) v4i*)g,
      (__attribute__((address_space(3))) v4i*)l, 0, 0);
#else
  *(uint4*)l = *(const uint4*)g;
#endif
}
__device__ __forceinline__ void cp_wait() {
#ifdef HAVE_ASYNC_LDS
  __builtin_amdgcn_s_wait_asynccnt(0);
#endif
}

// ---------------------------------------------------------------------------
// Kernel 1: fused QKV projection. grid = (M/128, D/128, 3), block = 256.
// Writes Q,K as [B,H,N,128] bf16 and V transposed [B,H,128,N] bf16.
// ---------------------------------------------------------------------------
__global__ __launch_bounds__(256) void qkv_gemm_kernel(
    const float* __restrict__ x,  const float* __restrict__ Wq,
    const float* __restrict__ Wk, const float* __restrict__ Wv,
    unsigned short* __restrict__ Qh, unsigned short* __restrict__ Kh,
    unsigned short* __restrict__ Vt) {
  __shared__ __align__(16) unsigned short ldsA[128 * 40];  // [m][k] k-pad 40
  __shared__ __align__(16) unsigned short ldsB[128 * 40];  // [n][k] transposed

  const int tid = threadIdx.x, lane = tid & 31, w = tid >> 5;
  const int half = lane >> 4, l16 = lane & 15;
  const int bm = blockIdx.x, bn = blockIdx.y, which = blockIdx.z;
  const float* W = (which == 0) ? Wq : (which == 1) ? Wk : Wv;
  const int wm = w & 3, wn = w >> 2;   // 4x2 wave grid -> 32x64 per wave

  FragC acc[2][4];
  for (int i = 0; i < 2; ++i)
    for (int j = 0; j < 4; ++j)
      for (int r = 0; r < 8; ++r) acc[i][j].f[r] = 0.f;

  for (int kt = 0; kt < DIM / 32; ++kt) {
    const int k0 = kt * 32;
    __syncthreads();
    // stage A: x[bm*128 .. +128)[k0 .. +32) -> bf16
#pragma unroll
    for (int i = 0; i < 4; ++i) {
      int t4 = tid + i * 256;                 // 1024 float4 chunks
      int row = t4 >> 3, c4 = (t4 & 7) << 2;
      float4 f = *(const float4*)&x[(size_t)(bm * 128 + row) * DIM + k0 + c4];
      uint2 p;
      p.x = (unsigned)f2bf(f.x) | ((unsigned)f2bf(f.y) << 16);
      p.y = (unsigned)f2bf(f.z) | ((unsigned)f2bf(f.w) << 16);
      *(uint2*)&ldsA[row * 40 + c4] = p;
    }
    // stage B transposed: W[k0..+32)[bn*128..+128) -> ldsB[n][k]
#pragma unroll
    for (int i = 0; i < 4; ++i) {
      int t4 = tid + i * 256;
      int row = t4 >> 5, c4 = (t4 & 31) << 2;
      float4 f = *(const float4*)&W[(size_t)(k0 + row) * DIM + bn * 128 + c4];
      ldsB[(c4 + 0) * 40 + row] = f2bf(f.x);
      ldsB[(c4 + 1) * 40 + row] = f2bf(f.y);
      ldsB[(c4 + 2) * 40 + row] = f2bf(f.z);
      ldsB[(c4 + 3) * 40 + row] = f2bf(f.w);
    }
    __syncthreads();

    FragAB a[2];
#pragma unroll
    for (int mt = 0; mt < 2; ++mt) {
      int r = wm * 32 + mt * 16 + l16;
#pragma unroll
      for (int v = 0; v < 8; ++v)
        a[mt].u[v] = *(const unsigned*)&ldsA[r * 40 + kmapA(v, half)];
    }
#pragma unroll
    for (int nt = 0; nt < 4; ++nt) {
      FragAB b;
      int n = wn * 64 + nt * 16 + l16;
#pragma unroll
      for (int v = 0; v < 8; ++v)
        b.u[v] = *(const unsigned*)&ldsB[n * 40 + 16 * half + 2 * v];
#pragma unroll
      for (int mt = 0; mt < 2; ++mt)
        acc[mt][nt].v = bf16_wmma(a[mt].v, b.v, acc[mt][nt].v);
    }
  }

  // scatter to head-major bf16 buffers (bn == head index since d_head == 128).
  // `which` is uniform (blockIdx.z): hoist it so the store loops are
  // straight-line (previous version emitted per-store branches).
  const int mrow0 = bm * 128 + wm * 32;         // + mt*16 + r + 8*half
  const int col0  = bn * 128 + wn * 64;         // + nt*16 + l16
  if (which == 2) {
#pragma unroll
    for (int mt = 0; mt < 2; ++mt)
#pragma unroll
      for (int nt = 0; nt < 4; ++nt)
#pragma unroll
        for (int r = 0; r < 8; ++r) {
          int m = mrow0 + mt * 16 + r + 8 * half;
          int c = col0 + nt * 16 + l16;
          int b = m >> 11, nq = m & (NSEQ - 1), dd = c & (DH - 1);
          Vt[((size_t)(b * NH + bn) * DH + dd) * NSEQ + nq] =
              f2bf(acc[mt][nt].f[r]);
        }
  } else {
    unsigned short* __restrict__ P = (which == 0) ? Qh : Kh;
#pragma unroll
    for (int mt = 0; mt < 2; ++mt)
#pragma unroll
      for (int nt = 0; nt < 4; ++nt)
#pragma unroll
        for (int r = 0; r < 8; ++r) {
          int m = mrow0 + mt * 16 + r + 8 * half;
          int c = col0 + nt * 16 + l16;
          int b = m >> 11, nq = m & (NSEQ - 1), dd = c & (DH - 1);
          P[((size_t)(b * NH + bn) * NSEQ + nq) * DH + dd] =
              f2bf(acc[mt][nt].f[r]);
        }
  }
}

// ---------------------------------------------------------------------------
// Kernel 2: flash attention. grid = (N/128, B*H), block = 256 (8 waves).
// Each wave owns 16 query rows end-to-end; K/V tiles of 64 keys via async LDS.
// ---------------------------------------------------------------------------
__global__ __launch_bounds__(256) void flash_attn_kernel(
    const unsigned short* __restrict__ Qh, const unsigned short* __restrict__ Kh,
    const unsigned short* __restrict__ Vt, float* __restrict__ out) {
  __shared__ __align__(16) unsigned short ldsK[64 * 136];  // [key][k]  pad 136
  __shared__ __align__(16) unsigned short ldsV[128 * 72];  // [dim][key] pad 72
  __shared__ __align__(16) unsigned short ldsP[128 * 72];  // [qrow][key] pad 72

  const int tid = threadIdx.x, lane = tid & 31, w = tid >> 5;
  const int half = lane >> 4, l16 = lane & 15;
  const int bh = blockIdx.y;             // b*16 + h
  const int q0 = blockIdx.x * 128;
  const size_t qkBase = (size_t)bh * NSEQ * DH;
  const size_t vBase  = (size_t)bh * DH * NSEQ;
  const float scale = 0.08838834764831845f;  // 1/sqrt(128)

  // resident Q A-fragments for this wave's 16 rows (4 k-steps of 32)
  FragAB aQ[4];
  {
    const unsigned short* qrow = &Qh[qkBase + (size_t)(q0 + w * 16 + l16) * DH];
#pragma unroll
    for (int ks = 0; ks < 4; ++ks)
#pragma unroll
      for (int v = 0; v < 8; ++v)
        aQ[ks].u[v] = *(const unsigned*)&qrow[ks * 32 + kmapA(v, half)];
  }

  FragC o[8];
  float m_i[8], l_i[8];
#pragma unroll
  for (int r = 0; r < 8; ++r) { m_i[r] = -__builtin_inff(); l_i[r] = 0.f; }
#pragma unroll
  for (int ot = 0; ot < 8; ++ot)
#pragma unroll
    for (int r = 0; r < 8; ++r) o[ot].f[r] = 0.f;

  for (int kt = 0; kt < NSEQ / 64; ++kt) {
    const int kb = kt * 64;
    __syncthreads();
    // stream K tile (64x128) and V tile (128x64) into LDS, 16B per op
#pragma unroll
    for (int i = 0; i < 4; ++i) {
      int c = tid + i * 256;
      int row = c >> 4, col8 = (c & 15) << 3;
      cp16(&Kh[qkBase + (size_t)(kb + row) * DH + col8], &ldsK[row * 136 + col8]);
    }
#pragma unroll
    for (int i = 0; i < 4; ++i) {
      int c = tid + i * 256;
      int row = c >> 3, col8 = (c & 7) << 3;
      cp16(&Vt[vBase + (size_t)row * NSEQ + kb + col8], &ldsV[row * 72 + col8]);
    }
    cp_wait();
    __syncthreads();

    // S = Q K^T  (16 rows x 64 keys per wave)
    FragC st[4];
#pragma unroll
    for (int nt = 0; nt < 4; ++nt) {
#pragma unroll
      for (int r = 0; r < 8; ++r) st[nt].f[r] = 0.f;
      int key = nt * 16 + l16;
#pragma unroll
      for (int ks = 0; ks < 4; ++ks) {
        FragAB b;
#pragma unroll
        for (int v = 0; v < 8; ++v)
          b.u[v] = *(const unsigned*)&ldsK[key * 136 + ks * 32 + 16 * half + 2 * v];
        st[nt].v = bf16_wmma(aQ[ks].v, b.v, st[nt].v);
      }
    }

    // online softmax, wave-local (row r lives on 16 lanes of one half)
#pragma unroll
    for (int r = 0; r < 8; ++r) {
      float s0 = st[0].f[r] * scale, s1 = st[1].f[r] * scale;
      float s2 = st[2].f[r] * scale, s3 = st[3].f[r] * scale;
      float mx = fmaxf(fmaxf(s0, s1), fmaxf(s2, s3));
#pragma unroll
      for (int off = 1; off < 16; off <<= 1)
        mx = fmaxf(mx, __shfl_xor(mx, off, 32));
      float mnew = fmaxf(m_i[r], mx);
      float alpha = __expf(m_i[r] - mnew);
      m_i[r] = mnew;
      float p0 = __expf(s0 - mnew), p1 = __expf(s1 - mnew);
      float p2 = __expf(s2 - mnew), p3 = __expf(s3 - mnew);
      float lsum = (p0 + p1) + (p2 + p3);
#pragma unroll
      for (int off = 1; off < 16; off <<= 1)
        lsum += __shfl_xor(lsum, off, 32);
      l_i[r] = l_i[r] * alpha + lsum;
      int prow = w * 16 + r + 8 * half;     // wave-private rows: no barrier
      ldsP[prow * 72 +  0 + l16] = f2bf(p0);
      ldsP[prow * 72 + 16 + l16] = f2bf(p1);
      ldsP[prow * 72 + 32 + l16] = f2bf(p2);
      ldsP[prow * 72 + 48 + l16] = f2bf(p3);
#pragma unroll
      for (int ot = 0; ot < 8; ++ot) o[ot].f[r] *= alpha;
    }
    __builtin_amdgcn_wave_barrier();  // LDS ops are in-order within a wave

    // reload P as A-fragments (C-layout -> A-layout via LDS)
    FragAB pA[2];
#pragma unroll
    for (int ks = 0; ks < 2; ++ks) {
      int prow = w * 16 + l16;
#pragma unroll
      for (int v = 0; v < 8; ++v)
        pA[ks].u[v] = *(const unsigned*)&ldsP[prow * 72 + ks * 32 + kmapA(v, half)];
    }
    // O += P V
#pragma unroll
    for (int ot = 0; ot < 8; ++ot) {
      int dim = ot * 16 + l16;
#pragma unroll
      for (int ks = 0; ks < 2; ++ks) {
        FragAB b;
#pragma unroll
        for (int v = 0; v < 8; ++v)
          b.u[v] = *(const unsigned*)&ldsV[dim * 72 + ks * 32 + 16 * half + 2 * v];
        o[ot].v = bf16_wmma(pA[ks].v, b.v, o[ot].v);
      }
    }
  }

  // epilogue: normalize and write fp32 [B,H,N,128]
#pragma unroll
  for (int ot = 0; ot < 8; ++ot)
#pragma unroll
    for (int r = 0; r < 8; ++r) {
      int row = q0 + w * 16 + r + 8 * half;
      int col = ot * 16 + l16;
      out[(size_t)bh * NSEQ * DH + (size_t)row * DH + col] = o[ot].f[r] / l_i[r];
    }
}

// ---------------------------------------------------------------------------
extern "C" void kernel_launch(void* const* d_in, const int* in_sizes, int n_in,
                              void* d_out, int out_size, void* d_ws, size_t ws_size,
                              hipStream_t stream) {
  (void)in_sizes; (void)n_in; (void)out_size; (void)ws_size;
  const float* x  = (const float*)d_in[0];
  const float* Wq = (const float*)d_in[1];
  const float* Wk = (const float*)d_in[2];
  const float* Wv = (const float*)d_in[3];
  float* out = (float*)d_out;

  const size_t elems = (size_t)NB * NSEQ * DIM;      // 16M bf16 per buffer
  unsigned short* Qh = (unsigned short*)d_ws;        // 32 MB
  unsigned short* Kh = Qh + elems;                   // 32 MB
  unsigned short* Vt = Kh + elems;                   // 32 MB (transposed V)

  dim3 g1((NB * NSEQ) / 128, DIM / 128, 3);
  qkv_gemm_kernel<<<g1, 256, 0, stream>>>(x, Wq, Wk, Wv, Qh, Kh, Vt);

  dim3 g2(NSEQ / 128, NB * NH);
  flash_attn_kernel<<<g2, 256, 0, stream>>>(Qh, Kh, Vt, out);
}
